// TGN_55559696941164
// MI455X (gfx1250) — matrix operand order repaired and checked by reading, not compiled, not run on hardware
//
#include <hip/hip_runtime.h>

#define EMB   128
#define HEADS 2
#define HD    64
#define NBR   20
#define CLS   10
#define KDIM  384   // 3*EMB
#define QDIM  256   // 2*EMB
#define NPW   16    // nodes per workgroup
#define NTHREADS 256

typedef __bf16 bf16x16 __attribute__((ext_vector_type(16)));
typedef float  f32x8   __attribute__((ext_vector_type(8)));
typedef unsigned short u16;
typedef unsigned int   u32;

__device__ __forceinline__ u16 f2bf(float f) {
  union { float f; u32 u; } v; v.f = f;
  u32 u = v.u + 0x7fffu + ((v.u >> 16) & 1u);   // round-to-nearest-even
  return (u16)(u >> 16);
}
__device__ __forceinline__ u32 pack2bf(float a, float b) {
  return (u32)f2bf(a) | ((u32)f2bf(b) << 16);
}
__device__ __forceinline__ float bf_lo(u32 p) { union { u32 u; float f; } v; v.u = p << 16;         return v.f; }
__device__ __forceinline__ float bf_hi(u32 p) { union { u32 u; float f; } v; v.u = p & 0xffff0000u; return v.f; }
__device__ __forceinline__ float bf1(u16 p)   { union { u32 u; float f; } v; v.u = ((u32)p) << 16;  return v.f; }

union FragU { uint4 q[2]; bf16x16 v; };

// A fragment, 16x32 bf16 from row-major LDS (rows 0..15 at given stride).
// ISA 7.12.2: lane l (l<16) row=l, elems 0..7 => K=c0..c0+7, elems 8..15 => K=c0+16..c0+23, c0 = kBase
//             lane l (>=16) row=l-16, same with c0 = kBase+8
__device__ __forceinline__ bf16x16 lds_a_frag(const u16* A, int stride, int kBase, int lane) {
  int r  = lane & 15;
  int c0 = kBase + ((lane & 16) ? 8 : 0);
  const u16* p = A + r * stride + c0;
  FragU u;
  u.q[0] = *reinterpret_cast<const uint4*>(p);
  u.q[1] = *reinterpret_cast<const uint4*>(p + 16);
  return u.v;
}
// B fragment, 32x16 bf16. Source BT is row-major with BT[n][k] = B[k][n] (weight rows are B columns).
// lane l: col n = nBase + (l&15); c0 = kBase + (l&16 ? 8:0); elems 0..7 => K = c0+e, elems 8..15 => K = c0+16+(e-8).
__device__ __forceinline__ bf16x16 glb_b_frag(const u16* BT, int stride, int nBase, int kBase, int lane) {
  int n = nBase + (lane & 15);
  int c = kBase + ((lane & 16) ? 8 : 0);
  const u16* p = BT + n * stride + c;
  FragU u;
  u.q[0] = *reinterpret_cast<const uint4*>(p);
  u.q[1] = *reinterpret_cast<const uint4*>(p + 16);
  return u.v;
}
__device__ __forceinline__ f32x8 wmma_bf16(bf16x16 a, bf16x16 b, f32x8 c) {
  return __builtin_amdgcn_wmma_f32_16x16x32_bf16(false, a, false, b, (short)0, c, false, false);
}

// ---------------- prep: bf16 weight images + folded Wco/bco + cos(tb) ----------------
__global__ void tgn_prep(const float* __restrict__ Wq, const float* __restrict__ Wk,
                         const float* __restrict__ Wv, const float* __restrict__ time_b,
                         const float* __restrict__ Wo, const float* __restrict__ Wc,
                         const float* __restrict__ bo, const float* __restrict__ bc,
                         u16* __restrict__ Wq_bf, u16* __restrict__ WkT_bf,
                         u16* __restrict__ Wv_bf, u16* __restrict__ qt_bf,
                         float* __restrict__ Wco, float* __restrict__ bco) {
  const int nWq = EMB * QDIM, nWk = KDIM * EMB, nWv = EMB * KDIM;
  const int nQt = EMB, nWco = CLS * EMB, nBco = CLS;
  const int total = nWq + nWk + nWv + nQt + nWco + nBco;
  int stride = gridDim.x * blockDim.x;
  for (int i = blockIdx.x * blockDim.x + threadIdx.x; i < total; i += stride) {
    int x = i;
    if (x < nWq) { Wq_bf[x] = f2bf(Wq[x]); continue; } x -= nWq;
    if (x < nWk) { int j = x / EMB, d = x % EMB; WkT_bf[x] = f2bf(Wk[d * KDIM + j]); continue; } x -= nWk;
    if (x < nWv) { Wv_bf[x] = f2bf(Wv[x]); continue; } x -= nWv;
    if (x < nQt) { qt_bf[x] = f2bf(cosf(time_b[x])); continue; } x -= nQt;
    if (x < nWco) {
      int c = x / EMB, e = x % EMB;
      float a = 0.f;
      for (int k = 0; k < EMB; ++k) a += Wc[c * EMB + k] * Wo[k * EMB + e];
      Wco[x] = a; continue;
    } x -= nWco;
    { int c = x; float a = bc[c];
      for (int k = 0; k < EMB; ++k) a += Wc[c * EMB + k] * bo[k];
      bco[c] = a; }
  }
}

// ---------------- conv: f32 embedding table -> bf16 table in workspace ----------------
__global__ void tgn_conv(const float* __restrict__ src, u16* __restrict__ dst, int n8) {
  int stride = gridDim.x * blockDim.x;
  for (int i = blockIdx.x * blockDim.x + threadIdx.x; i < n8; i += stride) {
    const float4* p = reinterpret_cast<const float4*>(src + (size_t)i * 8);
    float4 f0 = p[0], f1 = p[1];
    uint4 st;
    st.x = pack2bf(f0.x, f0.y); st.y = pack2bf(f0.z, f0.w);
    st.z = pack2bf(f1.x, f1.y); st.w = pack2bf(f1.z, f1.w);
    *reinterpret_cast<uint4*>(dst + (size_t)i * 8) = st;
  }
}

// ---------------- main fused kernel: 16 nodes per workgroup, 8 wave32 ----------------
__global__ __launch_bounds__(NTHREADS, 1)
void tgn_main(const int* __restrict__ node_data, const int* __restrict__ nbr_data,
              const unsigned char* __restrict__ nbr_mask,
              const float* __restrict__ v_emb, const float* __restrict__ r_emb,
              const float* __restrict__ time_w, const float* __restrict__ time_b,
              const float* __restrict__ bq, const float* __restrict__ bk,
              const float* __restrict__ bv, const float* __restrict__ bc,
              const u16* __restrict__ Wq_bf, const u16* __restrict__ WkT_bf,
              const u16* __restrict__ Wv_bf, const u16* __restrict__ qt_bf,
              const float* __restrict__ Wco, const float* __restrict__ bco,
              const u16* __restrict__ ve_bf, const u16* __restrict__ re_bf, int useBF,
              float* __restrict__ out, int N) {
  __shared__ alignas(16) u16 keyv[NPW * NBR * KDIM];     // 240 KB: [ne | re | cos(dt*tw+tb)] bf16
  __shared__ alignas(16) u16 qryA[NPW * QDIM];           // 8 KB
  __shared__ alignas(16) u16 qA[NPW * EMB];              // 4 KB
  __shared__ alignas(16) u16 uB[NPW * HEADS * KDIM];     // 24 KB   u = Wk_h^T q_h
  __shared__ alignas(16) u16 sA[HEADS * NPW * KDIM];     // 24 KB   s = sum att*keyv
  __shared__ float scoresL[NPW * HEADS * NBR];           // 2.5 KB (scores -> att in place)
  __shared__ float outO[NPW * EMB];                      // 8 KB
  __shared__ float cqL[NPW * HEADS];                     // (q_h . bk_h) * scale
  __shared__ float asumL[NPW * HEADS];                   // sum_nbr att
  __shared__ int nvL[NPW * NBR], nrL[NPW * NBR], ntL[NPW * NBR];
  __shared__ unsigned char mkL[NPW * NBR];
  __shared__ int tL[NPW], vidL[NPW], flagL[NPW], okL[NPW];
  __shared__ float twL[EMB], tbL[EMB];

  const int tid  = threadIdx.x;
  const int lane = tid & 31;
  const int wv   = tid >> 5;
  const int base = blockIdx.x * NPW;
  const float scale = 0.088388347648318447f;  // 1/sqrt(128)

  // ---- phase 0: metadata ----
  for (int idx = tid; idx < NPW; idx += NTHREADS) {
    int node = base + idx;
    int ok = node < N;
    okL[idx] = ok;
    int nn = ok ? node : 0;
    vidL[idx] = node_data[nn * 4 + 0];
    tL[idx]   = node_data[nn * 4 + 2];
  }
  for (int idx = tid; idx < NPW * NBR; idx += NTHREADS) {
    int i = idx / NBR, nb = idx % NBR;
    int node = base + i;
    int ok = node < N;
    int off = ((ok ? node : 0) * NBR + nb);
    nvL[idx] = nbr_data[off * 3 + 0];
    nrL[idx] = nbr_data[off * 3 + 1];
    ntL[idx] = nbr_data[off * 3 + 2];
    mkL[idx] = ok ? nbr_mask[off] : (unsigned char)1;
  }
  for (int idx = tid; idx < EMB; idx += NTHREADS) { twL[idx] = time_w[idx]; tbL[idx] = time_b[idx]; }
  __syncthreads();

  if (tid < NPW) {
    int f = 1;
    for (int nb = 0; nb < NBR; ++nb) f &= (mkL[tid * NBR + nb] != 0);
    flagL[tid] = f;
  }
  // ---- phase 1: build query rows and keyv, 8 elements (16B of bf16) per step ----
  for (int idx = tid; idx < NPW * (QDIM / 8); idx += NTHREADS) {
    int i = idx / (QDIM / 8);
    int c = (idx % (QDIM / 8)) * 8;
    uint4 st;
    if (c < EMB) {
      if (useBF) {
        st = *reinterpret_cast<const uint4*>(ve_bf + (size_t)vidL[i] * EMB + c);
      } else {
        const float4* p = reinterpret_cast<const float4*>(v_emb + (size_t)vidL[i] * EMB + c);
        float4 f0 = p[0], f1 = p[1];
        st.x = pack2bf(f0.x, f0.y); st.y = pack2bf(f0.z, f0.w);
        st.z = pack2bf(f1.x, f1.y); st.w = pack2bf(f1.z, f1.w);
      }
    } else {
      st = *reinterpret_cast<const uint4*>(qt_bf + (c - EMB));
    }
    *reinterpret_cast<uint4*>(qryA + i * QDIM + c) = st;
  }
  for (int idx = tid; idx < NPW * NBR * (KDIM / 8); idx += NTHREADS) {
    int row = idx / (KDIM / 8);
    int c = (idx % (KDIM / 8)) * 8;
    uint4 st;
    if (c < 2 * EMB) {
      int isR = c >= EMB;
      int src = isR ? nrL[row] : nvL[row];
      int cc  = isR ? (c - EMB) : c;
      if (useBF) {
        const u16* tab = isR ? re_bf : ve_bf;
        st = *reinterpret_cast<const uint4*>(tab + (size_t)src * EMB + cc);
      } else {
        const float* tab = isR ? r_emb : v_emb;
        const float4* p = reinterpret_cast<const float4*>(tab + (size_t)src * EMB + cc);
        float4 f0 = p[0], f1 = p[1];
        st.x = pack2bf(f0.x, f0.y); st.y = pack2bf(f0.z, f0.w);
        st.z = pack2bf(f1.x, f1.y); st.w = pack2bf(f1.z, f1.w);
      }
    } else {
      int i = row / NBR, j = c - 2 * EMB;
      float dt = (float)(tL[i] - ntL[row]);
      float cv[8];
      #pragma unroll
      for (int e = 0; e < 8; ++e) cv[e] = cosf(dt * twL[j + e] + tbL[j + e]);
      st.x = pack2bf(cv[0], cv[1]); st.y = pack2bf(cv[2], cv[3]);
      st.z = pack2bf(cv[4], cv[5]); st.w = pack2bf(cv[6], cv[7]);
    }
    *reinterpret_cast<uint4*>(keyv + row * KDIM + c) = st;
  }
  __syncthreads();

  // ---- phase 2: Q = query(16x256) @ Wq^T + bq  (8 waves x 8 K-steps of WMMA) ----
  {
    f32x8 acc = {0.f, 0.f, 0.f, 0.f, 0.f, 0.f, 0.f, 0.f};
    #pragma unroll
    for (int kc = 0; kc < QDIM / 32; ++kc) {
      bf16x16 a = lds_a_frag(qryA, QDIM, kc * 32, lane);
      bf16x16 b = glb_b_frag(Wq_bf, QDIM, wv * 16, kc * 32, lane);
      acc = wmma_bf16(a, b, acc);
    }
    int n = wv * 16 + (lane & 15);
    float bb = bq[n];
    int mb = (lane & 16) ? 8 : 0;
    #pragma unroll
    for (int e = 0; e < 8; ++e) qA[(mb + e) * EMB + n] = f2bf(acc[e] + bb);
  }
  __syncthreads();

  // ---- phase 3: per-(node,head) bk score offset, then U = q_h(16x64) @ Wk_h(64x384) ----
  if (tid < NPW * HEADS) {
    int i = tid >> 1, h = tid & 1;
    float a = 0.f;
    for (int d = 0; d < HD; ++d) a += bf1(qA[i * EMB + h * HD + d]) * bk[h * HD + d];
    cqL[tid] = a * scale;
  }
  for (int t = wv; t < 48; t += 8) {     // 2 heads x 24 column-tiles
    int h = t / 24, jt = t % 24;
    f32x8 acc = {0.f, 0.f, 0.f, 0.f, 0.f, 0.f, 0.f, 0.f};
    #pragma unroll
    for (int kc = 0; kc < 2; ++kc) {
      bf16x16 a = lds_a_frag(qA, EMB, h * HD + kc * 32, lane);
      bf16x16 b = glb_b_frag(WkT_bf, EMB, jt * 16, h * HD + kc * 32, lane);
      acc = wmma_bf16(a, b, acc);
    }
    int j = jt * 16 + (lane & 15);
    int mb = (lane & 16) ? 8 : 0;
    #pragma unroll
    for (int e = 0; e < 8; ++e) uB[((mb + e) * HEADS + h) * KDIM + j] = f2bf(acc[e]);
  }
  __syncthreads();

  // ---- phase 4: scores[i,nb,h] = (u . keyv)*scale + (q.bk)*scale  (ds_load_b128 dots) ----
  for (int idx = tid; idx < NPW * NBR * HEADS; idx += NTHREADS) {
    int i = idx / (NBR * HEADS), rem = idx % (NBR * HEADS);
    int nb = rem >> 1, h = rem & 1;
    float s;
    if (mkL[i * NBR + nb]) s = -1.0e9f;
    else {
      const uint4* up = reinterpret_cast<const uint4*>(uB + (i * HEADS + h) * KDIM);
      const uint4* kp = reinterpret_cast<const uint4*>(keyv + (i * NBR + nb) * KDIM);
      float a = 0.f;
      for (int j = 0; j < KDIM / 8; ++j) {
        uint4 x = up[j], y = kp[j];
        a += bf_lo(x.x) * bf_lo(y.x) + bf_hi(x.x) * bf_hi(y.x);
        a += bf_lo(x.y) * bf_lo(y.y) + bf_hi(x.y) * bf_hi(y.y);
        a += bf_lo(x.z) * bf_lo(y.z) + bf_hi(x.z) * bf_hi(y.z);
        a += bf_lo(x.w) * bf_lo(y.w) + bf_hi(x.w) * bf_hi(y.w);
      }
      s = a * scale + cqL[i * HEADS + h];
    }
    scoresL[(i * HEADS + h) * NBR + nb] = s;
  }
  __syncthreads();

  // ---- phase 5: softmax over the HEADS axis (matches reference's axis=-1) ----
  for (int idx = tid; idx < NPW * NBR; idx += NTHREADS) {
    int i = idx / NBR, nb = idx % NBR;
    float s0 = scoresL[(i * HEADS + 0) * NBR + nb];
    float s1 = scoresL[(i * HEADS + 1) * NBR + nb];
    float m = fmaxf(s0, s1);
    float e0 = __expf(s0 - m), e1 = __expf(s1 - m);
    float inv = 1.f / (e0 + e1);
    scoresL[(i * HEADS + 0) * NBR + nb] = e0 * inv;
    scoresL[(i * HEADS + 1) * NBR + nb] = e1 * inv;
  }
  __syncthreads();

  // ---- phase 6: att-sum per (i,h) and s_h = sum_nb att * keyv (2 outputs per step) ----
  if (tid < NPW * HEADS) {
    float a = 0.f;
    for (int nb = 0; nb < NBR; ++nb) a += scoresL[tid * NBR + nb];
    asumL[tid] = a;
  }
  for (int idx = tid; idx < HEADS * NPW * (KDIM / 2); idx += NTHREADS) {
    int h = idx / (NPW * (KDIM / 2)), rem = idx % (NPW * (KDIM / 2));
    int i = rem / (KDIM / 2), jp = rem % (KDIM / 2);
    const float* ap = scoresL + (i * HEADS + h) * NBR;
    float a0 = 0.f, a1 = 0.f;
    for (int nb = 0; nb < NBR; ++nb) {
      u32 kv = reinterpret_cast<const u32*>(keyv + (i * NBR + nb) * KDIM)[jp];
      float w = ap[nb];
      a0 += w * bf_lo(kv);
      a1 += w * bf_hi(kv);
    }
    reinterpret_cast<u32*>(sA)[(h * NPW + i) * (KDIM / 2) + jp] = pack2bf(a0, a1);
  }
  __syncthreads();

  // ---- phase 7: out_h = s_h(16x384) @ Wv_h^T + asum*bv  (8 waves x 12 K-steps WMMA) ----
  {
    int h = wv >> 2, nt = wv & 3;
    f32x8 acc = {0.f, 0.f, 0.f, 0.f, 0.f, 0.f, 0.f, 0.f};
    #pragma unroll
    for (int kc = 0; kc < KDIM / 32; ++kc) {
      bf16x16 a = lds_a_frag(sA + h * NPW * KDIM, KDIM, kc * 32, lane);
      bf16x16 b = glb_b_frag(Wv_bf, KDIM, h * HD + nt * 16, kc * 32, lane);
      acc = wmma_bf16(a, b, acc);
    }
    int col = h * HD + nt * 16 + (lane & 15);
    float bvv = bv[col];
    int mb = (lane & 16) ? 8 : 0;
    #pragma unroll
    for (int e = 0; e < 8; ++e) {
      int m = mb + e;
      outO[m * EMB + col] = acc[e] + asumL[m * HEADS + h] * bvv;
    }
  }
  __syncthreads();

  // ---- phase 8: final = out @ (Wc Wo)^T + (Wc bo + bc); all-masked nodes -> bc ----
  for (int idx = tid; idx < NPW * CLS; idx += NTHREADS) {
    int i = idx / CLS, c = idx % CLS;
    if (!okL[i]) continue;
    float r;
    if (flagL[i]) r = bc[c];
    else {
      float a = bco[c];
      const float* wp = Wco + c * EMB;
      const float* op = outO + i * EMB;
      for (int e = 0; e < EMB; ++e) a += op[e] * wp[e];
      r = a;
    }
    out[(base + i) * CLS + c] = r;
  }
}

extern "C" void kernel_launch(void* const* d_in, const int* in_sizes, int n_in,
                              void* d_out, int out_size, void* d_ws, size_t ws_size,
                              hipStream_t stream) {
  (void)n_in; (void)out_size;
  const int*   node_data = (const int*)d_in[0];
  const int*   nbr_data  = (const int*)d_in[1];
  const unsigned char* nbr_mask = (const unsigned char*)d_in[2];  // jnp bool = 1 byte
  const float* v_emb  = (const float*)d_in[3];
  const float* r_emb  = (const float*)d_in[4];
  const float* time_w = (const float*)d_in[5];
  const float* time_b = (const float*)d_in[6];
  const float* Wq = (const float*)d_in[7];
  const float* bq = (const float*)d_in[8];
  const float* Wk = (const float*)d_in[9];
  const float* bk = (const float*)d_in[10];
  const float* Wv = (const float*)d_in[11];
  const float* bv = (const float*)d_in[12];
  const float* Wo = (const float*)d_in[13];
  const float* bo = (const float*)d_in[14];
  const float* Wc = (const float*)d_in[15];
  const float* bc = (const float*)d_in[16];
  const int N = in_sizes[0] / 4;

  char* ws = (char*)d_ws;
  u16*   Wq_bf  = (u16*)(ws + 0);                         // 128*256 bf16
  u16*   WkT_bf = (u16*)(ws + 65536);                     // 384*128 bf16 (transposed)
  u16*   Wv_bf  = (u16*)(ws + 65536 + 98304);             // 128*384 bf16
  u16*   qt_bf  = (u16*)(ws + 262144);                    // cos(time_b) bf16
  float* Wco    = (float*)(ws + 262400);                  // 10x128 folded Wc@Wo
  float* bco    = (float*)(ws + 267520);                  // 10 folded Wc@bo+bc

  // optional bf16 embedding tables (halves gather bytes; tables become L2-resident)
  size_t nV = (size_t)in_sizes[3] / EMB;
  size_t nR = (size_t)in_sizes[4] / EMB;
  size_t vOff = 270336;                                   // 4KB-aligned past fixed region
  size_t rOff = vOff + nV * EMB * sizeof(u16);
  size_t need = rOff + nR * EMB * sizeof(u16);
  int useBF = (ws_size >= need) ? 1 : 0;
  u16* ve_bf = (u16*)(ws + vOff);
  u16* re_bf = (u16*)(ws + rOff);

  tgn_prep<<<256, NTHREADS, 0, stream>>>(Wq, Wk, Wv, time_b, Wo, Wc, bo, bc,
                                         Wq_bf, WkT_bf, Wv_bf, qt_bf, Wco, bco);
  if (useBF) {
    tgn_conv<<<2048, NTHREADS, 0, stream>>>(v_emb, ve_bf, (int)(nV * EMB / 8));
    tgn_conv<<<2048, NTHREADS, 0, stream>>>(r_emb, re_bf, (int)(nR * EMB / 8));
  }
  int ngrid = (N + NPW - 1) / NPW;
  tgn_main<<<ngrid, NTHREADS, 0, stream>>>(node_data, nbr_data, nbr_mask, v_emb, r_emb,
                                           time_w, time_b, bq, bk, bv, bc,
                                           Wq_bf, WkT_bf, Wv_bf, qt_bf, Wco, bco,
                                           ve_bf, re_bf, useBF,
                                           (float*)d_out, N);
}
